// CorrLayer_bidcos_78426102825595
// MI455X (gfx1250) — matrix operand order repaired and checked by reading, not compiled, not run on hardware
//
#include <hip/hip_runtime.h>
#include <hip/hip_bf16.h>
#include <stdint.h>

#define BB 2
#define NN 4096
#define CCH 128
#define KNBR 16
#define NKTOT (NN * KNBR)   // 65536

typedef _Float16 half8  __attribute__((ext_vector_type(8)));
typedef _Float16 half16 __attribute__((ext_vector_type(16)));
typedef float    float8 __attribute__((ext_vector_type(8)));

// ---------------------------------------------------------------- helpers ---

__device__ inline half16 pack16(half8 lo, half8 hi) {
    half16 r;
#pragma unroll
    for (int i = 0; i < 8; ++i) { r[i] = lo[i]; r[i + 8] = hi[i]; }
    return r;
}

__device__ inline float8 zero8() {
    float8 z;
#pragma unroll
    for (int i = 0; i < 8; ++i) z[i] = 0.f;
    return z;
}

// A-matrix fragment (16x32 f16, MxK). ISA layout: lane = 16*half + m holds
// row M=m, K runs [8*half .. 8*half+7] and [16+8*half .. 23+8*half] (+kbase).
__device__ inline half16 load_frag_a(const _Float16* base, int row0, int stride, int kbase) {
    int lane = threadIdx.x & 31;
    int hs = lane >> 4;
    int m = lane & 15;
    const _Float16* p = base + (size_t)(row0 + m) * stride + kbase + hs * 8;
    half8 lo = *(const half8*)(p);
    half8 hi = *(const half8*)(p + 16);
    return pack16(lo, hi);
}

// B-matrix fragment (32x16 f16, KxN). Same K order as A: lane = 16*half + n holds
// K [8*half .. 8*half+7] and [16+8*half .. 23+8*half] for column N=lane&15.
__device__ inline half16 load_frag_b(const _Float16* base, int row0, int stride, int kbase) {
    int lane = threadIdx.x & 31;
    int hs = lane >> 4;
    int m = lane & 15;
    const _Float16* p = base + (size_t)(row0 + m) * stride + kbase + hs * 8;
    half8 lo = *(const half8*)(p);
    half8 hi = *(const half8*)(p + 16);
    return pack16(lo, hi);
}

template <int NTOP>
__device__ inline void topk_insert(float (&d)[NTOP], int (&id)[NTOP], float v, int i) {
    if (v >= d[NTOP - 1]) return;
    int p = NTOP - 1;
#pragma unroll
    for (int q = NTOP - 1; q >= 1; --q) {
        if (d[q - 1] > v) { d[q] = d[q - 1]; id[q] = id[q - 1]; p = q - 1; }
    }
    d[p] = v; id[p] = i;
}

// --------------------------------------------------------------- kernels ----

// Pack all weights to f16 layouts used by the WMMA GEMMs.
__global__ void prep_w_kernel(const float* __restrict__ t11, const float* __restrict__ t22,
                              const float* __restrict__ mlp, const float* __restrict__ fe1,
                              const float* __restrict__ fe2,
                              _Float16* __restrict__ wt11, _Float16* __restrict__ wt22,
                              _Float16* __restrict__ wmlp, _Float16* __restrict__ wfe2,
                              _Float16* __restrict__ wf1a, _Float16* __restrict__ wf1b) {
    int gid = blockIdx.x * 256 + threadIdx.x;
    int stride = gridDim.x * 256;
    for (int i = gid; i < 128 * 128; i += stride) {
        wt11[i] = (_Float16)t11[i];
        wt22[i] = (_Float16)t22[i];
        wmlp[i] = (_Float16)mlp[i];
        wfe2[i] = (_Float16)fe2[i];
        int o = i >> 7, c = i & 127;
        wf1a[i] = (_Float16)fe1[o * 259 + c];          // f1 block of fe_w1
    }
    for (int i = gid; i < 128 * 160; i += stride) {
        int o = i / 160, c = i - o * 160;
        float v = 0.f;
        if (c < 128) v = fe1[o * 259 + 128 + c];       // gathered-feat block
        else if (c < 131) v = fe1[o * 259 + 256 + (c - 128)];  // dxyz block
        wf1b[i] = (_Float16)v;
    }
}

// L2-normalize features along C, write row-major f16 (B,N,128).
__global__ __launch_bounds__(256) void normfeat_kernel(const float* __restrict__ X,
                                                       _Float16* __restrict__ Yh) {
    int g = blockIdx.x * 256 + threadIdx.x;
    int b = g / NN, n = g - b * NN;
    float s = 0.f;
    for (int c = 0; c < 128; ++c) {
        float v = X[((size_t)b * 128 + c) * NN + n];
        s += v * v;
    }
    float inv = 1.f / fmaxf(sqrtf(s), 1e-12f);
    _Float16* yr = Yh + ((size_t)b * NN + n) * 128;
    for (int c = 0; c < 128; ++c) {
        float v = X[((size_t)b * 128 + c) * NN + n];
        yr[c] = (_Float16)(v * inv);
    }
}

// (B,C,N) fp32 -> (B,N,C) f16 transpose.
__global__ void transpose_kernel(const float* __restrict__ X, _Float16* __restrict__ Y) {
    __shared__ float tile[32][33];
    int bn = blockIdx.x * 32, bc = blockIdx.y * 32, b = blockIdx.z;
    for (int i = threadIdx.y; i < 32; i += 8)
        tile[i][threadIdx.x] = X[((size_t)b * 128 + bc + i) * NN + bn + threadIdx.x];
    __syncthreads();
    for (int i = threadIdx.y; i < 32; i += 8)
        Y[((size_t)b * NN + bn + i) * 128 + bc + threadIdx.x] = (_Float16)tile[threadIdx.x][i];
}

// PointPWC warp: 3-NN of each xyz2 point among xyz1+flow, inverse-distance blend.
__global__ __launch_bounds__(256) void warp_kernel(const float* __restrict__ xyz1,
                                                   const float* __restrict__ xyz2,
                                                   const float* __restrict__ flow,
                                                   float* __restrict__ warped) {
    __shared__ float sx[256], sy[256], sz[256];
    const int b = blockIdx.y;
    const int n2 = blockIdx.x * 256 + threadIdx.x;
    float px = xyz2[(size_t)(b * 3 + 0) * NN + n2];
    float py = xyz2[(size_t)(b * 3 + 1) * NN + n2];
    float pz = xyz2[(size_t)(b * 3 + 2) * NN + n2];
    float d[3]; int id[3];
#pragma unroll
    for (int q = 0; q < 3; ++q) { d[q] = 3.0e38f; id[q] = 0; }
    for (int c0 = 0; c0 < NN; c0 += 256) {
        __syncthreads();
        sx[threadIdx.x] = xyz1[(size_t)(b * 3 + 0) * NN + c0 + threadIdx.x] +
                          flow[(size_t)(b * 3 + 0) * NN + c0 + threadIdx.x];
        sy[threadIdx.x] = xyz1[(size_t)(b * 3 + 1) * NN + c0 + threadIdx.x] +
                          flow[(size_t)(b * 3 + 1) * NN + c0 + threadIdx.x];
        sz[threadIdx.x] = xyz1[(size_t)(b * 3 + 2) * NN + c0 + threadIdx.x] +
                          flow[(size_t)(b * 3 + 2) * NN + c0 + threadIdx.x];
        __syncthreads();
        for (int j = 0; j < 256; ++j) {
            float dx = px - sx[j], dy = py - sy[j], dz = pz - sz[j];
            topk_insert<3>(d, id, dx * dx + dy * dy + dz * dz, c0 + j);
        }
    }
    float w[3], wsum = 0.f;
#pragma unroll
    for (int q = 0; q < 3; ++q) {
        float dd = fmaxf(sqrtf(fmaxf(d[q], 1e-12f)), 1e-10f);
        w[q] = 1.f / dd; wsum += w[q];
    }
    float fx = 0.f, fy = 0.f, fz = 0.f;
#pragma unroll
    for (int q = 0; q < 3; ++q) {
        float wq = w[q] / wsum;
        fx += wq * flow[(size_t)(b * 3 + 0) * NN + id[q]];
        fy += wq * flow[(size_t)(b * 3 + 1) * NN + id[q]];
        fz += wq * flow[(size_t)(b * 3 + 2) * NN + id[q]];
    }
    warped[(size_t)(b * 3 + 0) * NN + n2] = px - fx;
    warped[(size_t)(b * 3 + 1) * NN + n2] = py - fy;
    warped[(size_t)(b * 3 + 2) * NN + n2] = pz - fz;
}

// kNN. MODE 0 (cross): top-8 by point distance + top-8 by feature cos distance
// (cosine tile computed with v_wmma_f32_16x16x32_f16 on normalized f16 feats).
// MODE 1 (flow_embed): top-16 by point distance only. idxOut (B,N,16).
template <int MODE>
__global__ __launch_bounds__(256) void knn_kernel(const float* __restrict__ x1,
                                                  const float* __restrict__ x2,
                                                  const _Float16* __restrict__ nf1,
                                                  const _Float16* __restrict__ nf2,
                                                  int* __restrict__ idxOut) {
    const int b = blockIdx.y;
    const int n1base = blockIdx.x * 64;
    const int tid = threadIdx.x;
    const int lane = tid & 31, wv = tid >> 5;

    __shared__ float cxs[64], cys[64], czs[64];
    __shared__ float simtile[(MODE == 0) ? 64 * 68 : 64];

    constexpr int NP = (MODE == 0) ? 8 : 16;
    float dP[NP]; int iP[NP];
    float dF[8];  int iF[8];
#pragma unroll
    for (int q = 0; q < NP; ++q) { dP[q] = 3.0e38f; iP[q] = 0; }
#pragma unroll
    for (int q = 0; q < 8; ++q) { dF[q] = 3.0e38f; iF[q] = 0; }

    float rx = 0.f, ry = 0.f, rz = 0.f;
    if (tid < 64) {
        int n1 = n1base + tid;
        rx = x1[(size_t)(b * 3 + 0) * NN + n1];
        ry = x1[(size_t)(b * 3 + 1) * NN + n1];
        rz = x1[(size_t)(b * 3 + 2) * NN + n1];
    }
    const int mt = wv >> 1;
    const int nt0 = (wv & 1) * 2, nt1 = nt0 + 1;
    const int hs = lane >> 4, m = lane & 15;

    for (int c0 = 0; c0 < NN; c0 += 64) {
        if (tid < 64) {
            cxs[tid] = x2[(size_t)(b * 3 + 0) * NN + c0 + tid];
            cys[tid] = x2[(size_t)(b * 3 + 1) * NN + c0 + tid];
            czs[tid] = x2[(size_t)(b * 3 + 2) * NN + c0 + tid];
        }
        if (MODE == 0) {
            float8 acc0 = zero8(), acc1 = zero8();
            const _Float16* abase = nf1 + (size_t)b * NN * 128;
            const _Float16* bbase = nf2 + (size_t)b * NN * 128;
#pragma unroll
            for (int ks = 0; ks < 4; ++ks) {
                half16 af = load_frag_a(abase, n1base + mt * 16, 128, ks * 32);
                half16 bf0 = load_frag_b(bbase, c0 + nt0 * 16, 128, ks * 32);
                half16 bf1 = load_frag_b(bbase, c0 + nt1 * 16, 128, ks * 32);
                acc0 = __builtin_amdgcn_wmma_f32_16x16x32_f16(false, af, false, bf0,
                                                              (short)0, acc0, false, false);
                acc1 = __builtin_amdgcn_wmma_f32_16x16x32_f16(false, af, false, bf1,
                                                              (short)0, acc1, false, false);
            }
#pragma unroll
            for (int r = 0; r < 8; ++r) {
                int rl = mt * 16 + r + 8 * hs;
                simtile[rl * 68 + nt0 * 16 + m] = 1.0f - acc0[r];
                simtile[rl * 68 + nt1 * 16 + m] = 1.0f - acc1[r];
            }
        }
        __syncthreads();
        if (tid < 64) {
            for (int j = 0; j < 64; ++j) {
                float dx = rx - cxs[j], dy = ry - cys[j], dz = rz - czs[j];
                topk_insert<NP>(dP, iP, dx * dx + dy * dy + dz * dz, c0 + j);
                if (MODE == 0)
                    topk_insert<8>(dF, iF, simtile[tid * 68 + j], c0 + j);
            }
        }
        __syncthreads();
    }
    if (tid < 64) {
        int* op = idxOut + ((size_t)b * NN + n1base + tid) * 16;
        if (MODE == 0) {
#pragma unroll
            for (int q = 0; q < 8; ++q) { op[q] = iP[q]; op[8 + q] = iF[q]; }
        } else {
#pragma unroll
            for (int q = 0; q < 16; ++q) op[q] = iP[q];
        }
    }
}

// cross builder: T[b, n*16+k, c] = p2[idx] + p1[n] + posw*dxyz + posb (f16 out),
// plus per-(b,group) sum/sumsq via wave32 shfl reduce + f32 atomics.
__global__ __launch_bounds__(128) void build_cross_kernel(
        const float* __restrict__ x1, const float* __restrict__ x2,
        const _Float16* __restrict__ p1r, const _Float16* __restrict__ p2r,
        const int* __restrict__ idx, const float* __restrict__ posw,
        const float* __restrict__ posb, _Float16* __restrict__ T,
        float* __restrict__ stats) {
    __shared__ int lid[128];
    __shared__ float ldx[128][3];
    const int b = blockIdx.y;
    const int n0 = blockIdx.x * 8;
    const int c = threadIdx.x;
    {
        int p = c;
        int n = n0 + (p >> 4), k = p & 15;
        int id = idx[((size_t)b * NN + n) * 16 + k];
        lid[p] = id;
        ldx[p][0] = x2[(size_t)(b * 3 + 0) * NN + id] - x1[(size_t)(b * 3 + 0) * NN + n];
        ldx[p][1] = x2[(size_t)(b * 3 + 1) * NN + id] - x1[(size_t)(b * 3 + 1) * NN + n];
        ldx[p][2] = x2[(size_t)(b * 3 + 2) * NN + id] - x1[(size_t)(b * 3 + 2) * NN + n];
    }
    __syncthreads();
    float w0 = posw[c * 3 + 0], w1 = posw[c * 3 + 1], w2 = posw[c * 3 + 2];
    float pb = posb[c];
    float sum = 0.f, ss = 0.f;
    for (int p = 0; p < 128; ++p) {
        int n = n0 + (p >> 4);
        int id = lid[p];
        float v = (float)p2r[((size_t)b * NN + id) * 128 + c] +
                  (float)p1r[((size_t)b * NN + n) * 128 + c] +
                  w0 * ldx[p][0] + w1 * ldx[p][1] + w2 * ldx[p][2] + pb;
        _Float16 vh = (_Float16)v;
        T[((size_t)b * NKTOT + n0 * 16 + p) * 128 + c] = vh;
        float vq = (float)vh;   // stats consistent with stored f16 value
        sum += vq; ss += vq * vq;
    }
#pragma unroll
    for (int off = 8; off >= 1; off >>= 1) {
        sum += __shfl_xor(sum, off, 32);
        ss  += __shfl_xor(ss,  off, 32);
    }
    if ((c & 15) == 0) {
        atomicAdd(&stats[((size_t)b * 8 + (c >> 4)) * 2 + 0], sum);
        atomicAdd(&stats[((size_t)b * 8 + (c >> 4)) * 2 + 1], ss);
    }
}

// flow builder: T[b, n*16+k, 0:160] = [gathered f2n (128) | dxyz (3) | zeros]
__global__ __launch_bounds__(160) void build_flow_kernel(
        const float* __restrict__ x1, const float* __restrict__ x2,
        const _Float16* __restrict__ f2r, const int* __restrict__ idx,
        _Float16* __restrict__ T) {
    __shared__ int lid[128];
    __shared__ float ldx[128][3];
    const int b = blockIdx.y;
    const int n0 = blockIdx.x * 8;
    const int c = threadIdx.x;
    if (c < 128) {
        int p = c;
        int n = n0 + (p >> 4), k = p & 15;
        int id = idx[((size_t)b * NN + n) * 16 + k];
        lid[p] = id;
        ldx[p][0] = x2[(size_t)(b * 3 + 0) * NN + id] - x1[(size_t)(b * 3 + 0) * NN + n];
        ldx[p][1] = x2[(size_t)(b * 3 + 1) * NN + id] - x1[(size_t)(b * 3 + 1) * NN + n];
        ldx[p][2] = x2[(size_t)(b * 3 + 2) * NN + id] - x1[(size_t)(b * 3 + 2) * NN + n];
    }
    __syncthreads();
    for (int p = 0; p < 128; ++p) {
        int id = lid[p];
        float v;
        if (c < 128) v = (float)f2r[((size_t)b * NN + id) * 128 + c];
        else if (c < 131) v = ldx[p][c - 128];
        else v = 0.f;
        T[((size_t)b * NKTOT + n0 * 16 + p) * 160 + c] = (_Float16)v;
    }
}

// Generic WMMA GEMM: Y[b,col,o] = sum_c W[o,c]*act(X[b,col,c]) + bias[o] (+AF).
// act: leaky(x*aff_scale + aff_shift) when aff != null (fused GroupNorm+leaky).
// addf: k-invariant field (B, NCOL/16, 128) added per output (for 259-ch conv).
// statsOut: per-(b, group=o/16) sum/sumsq accumulation for the next GroupNorm.
// All feature tensors f16 (bandwidth-bound pipeline); accumulation/stats f32.
template <int CINP>
__global__ __launch_bounds__(256) void gconv_kernel(
        const _Float16* __restrict__ X, int NCOL,
        const _Float16* __restrict__ W, const float* __restrict__ bias,
        const float* __restrict__ aff, const _Float16* __restrict__ addf,
        _Float16* __restrict__ Y, float* __restrict__ statsOut) {
    constexpr int LSTR = CINP + 8;
    constexpr int SBIN = 64 * LSTR * 2;
    constexpr int SBOUT = 64 * 129 * 4;
    constexpr int SBYTES = (SBIN > SBOUT) ? SBIN : SBOUT;
    __shared__ __align__(16) unsigned char ldsraw[SBYTES];
    _Float16* XT = reinterpret_cast<_Float16*>(ldsraw);
    float* OT = reinterpret_cast<float*>(ldsraw);

    const int b = blockIdx.y;
    const int colbase = blockIdx.x * 64;
    const int tid = threadIdx.x;
    const int wv = tid >> 5, lane = tid & 31;
    const int hs = lane >> 4, m = lane & 15;

    for (int i = tid; i < 64 * CINP; i += 256) {
        int r = i / CINP, c = i - r * CINP;
        float x = (float)X[((size_t)b * NCOL + colbase + r) * CINP + c];
        if (aff) {
            float sc = aff[(b * 128 + c) * 2 + 0];
            float sh = aff[(b * 128 + c) * 2 + 1];
            x = x * sc + sh;
            x = (x > 0.f) ? x : 0.1f * x;
        }
        XT[r * LSTR + c] = (_Float16)x;
    }
    __syncthreads();

    float8 acc[4];
#pragma unroll
    for (int s = 0; s < 4; ++s) acc[s] = zero8();

    const _Float16* wrow = W + (size_t)(wv * 16) * CINP;
#pragma unroll
    for (int ks = 0; ks < CINP / 32; ++ks) {
        half16 af = load_frag_a(wrow, 0, CINP, ks * 32);
        // hoist all B fragments so ds_loads clause together ahead of the WMMAs
        half16 bf0 = load_frag_b(XT,  0, LSTR, ks * 32);
        half16 bf1 = load_frag_b(XT, 16, LSTR, ks * 32);
        half16 bf2 = load_frag_b(XT, 32, LSTR, ks * 32);
        half16 bf3 = load_frag_b(XT, 48, LSTR, ks * 32);
        acc[0] = __builtin_amdgcn_wmma_f32_16x16x32_f16(false, af, false, bf0,
                                                        (short)0, acc[0], false, false);
        acc[1] = __builtin_amdgcn_wmma_f32_16x16x32_f16(false, af, false, bf1,
                                                        (short)0, acc[1], false, false);
        acc[2] = __builtin_amdgcn_wmma_f32_16x16x32_f16(false, af, false, bf2,
                                                        (short)0, acc[2], false, false);
        acc[3] = __builtin_amdgcn_wmma_f32_16x16x32_f16(false, af, false, bf3,
                                                        (short)0, acc[3], false, false);
    }
    __syncthreads();

    float lsum = 0.f, lss = 0.f;
#pragma unroll
    for (int s = 0; s < 4; ++s) {
        int cl = s * 16 + m;
#pragma unroll
        for (int r = 0; r < 8; ++r) {
            int o = wv * 16 + r + 8 * hs;
            float v = acc[s][r] + (bias ? bias[o] : 0.f);
            if (addf) v += (float)addf[((size_t)b * (NCOL >> 4) + ((colbase + cl) >> 4)) * 128 + o];
            lsum += v; lss += v * v;
            OT[cl * 129 + o] = v;
        }
    }
    if (statsOut) {
#pragma unroll
        for (int off = 16; off >= 1; off >>= 1) {
            lsum += __shfl_xor(lsum, off, 32);
            lss  += __shfl_xor(lss,  off, 32);
        }
        if (lane == 0) {
            atomicAdd(&statsOut[((size_t)b * 8 + wv) * 2 + 0], lsum);
            atomicAdd(&statsOut[((size_t)b * 8 + wv) * 2 + 1], lss);
        }
    }
    __syncthreads();
    for (int i = tid; i < 64 * 128; i += 256) {
        int r = i >> 7, o = i & 127;
        Y[((size_t)b * NCOL + colbase + r) * 128 + o] = (_Float16)OT[r * 129 + o];
    }
}

// GroupNorm stats (sum,sumsq) -> per-(b,channel) affine (scale,shift).
__global__ void affine_kernel(const float* __restrict__ stats,
                              const float* __restrict__ gam,
                              const float* __restrict__ bet,
                              float cnt, float* __restrict__ aff) {
    int t = threadIdx.x;
    if (t < BB * 128) {
        int b = t >> 7, c = t & 127, g = c >> 4;
        float s  = stats[((size_t)b * 8 + g) * 2 + 0];
        float s2 = stats[((size_t)b * 8 + g) * 2 + 1];
        float mean = s / cnt;
        float var = s2 / cnt - mean * mean;
        float rs = rsqrtf(var + 1e-5f);
        float sc = rs * gam[c];
        aff[t * 2 + 0] = sc;
        aff[t * 2 + 1] = bet[c] - mean * sc;
    }
}

// leaky(gn(u)) then max over K neighbors. mode 0: f16 (B,N,128) row-major;
// mode 1: f32 (B,128,N) channel-major (final output).
__global__ __launch_bounds__(128) void finalize_kernel(const _Float16* __restrict__ U,
                                                       const float* __restrict__ aff,
                                                       _Float16* __restrict__ outH,
                                                       float* __restrict__ outF,
                                                       int flowmode) {
    int b = blockIdx.y, n = blockIdx.x, o = threadIdx.x;
    float sc = aff[(b * 128 + o) * 2 + 0];
    float sh = aff[(b * 128 + o) * 2 + 1];
    float mx = -3.0e38f;
    for (int k = 0; k < 16; ++k) {
        float v = (float)U[((size_t)b * NKTOT + n * 16 + k) * 128 + o] * sc + sh;
        v = (v > 0.f) ? v : 0.1f * v;
        mx = fmaxf(mx, v);
    }
    if (flowmode) outF[((size_t)b * 128 + o) * NN + n] = mx;
    else          outH[((size_t)b * NN + n) * 128 + o] = (_Float16)mx;
}

// ---------------------------------------------------------------- driver ----

extern "C" void kernel_launch(void* const* d_in, const int* in_sizes, int n_in,
                              void* d_out, int out_size, void* d_ws, size_t ws_size,
                              hipStream_t stream) {
    (void)in_sizes; (void)n_in; (void)out_size; (void)ws_size;
    const float* xyz1   = (const float*)d_in[0];
    const float* xyz2   = (const float*)d_in[1];
    const float* feat1  = (const float*)d_in[2];
    const float* feat2  = (const float*)d_in[3];
    const float* flow   = (const float*)d_in[4];
    const float* t11_w  = (const float*)d_in[5];
    const float* t11_b  = (const float*)d_in[6];
    const float* t22_w  = (const float*)d_in[7];
    const float* t22_b  = (const float*)d_in[8];
    const float* pos_w  = (const float*)d_in[9];
    const float* pos_b  = (const float*)d_in[10];
    const float* gn_g   = (const float*)d_in[11];
    const float* gn_b   = (const float*)d_in[12];
    const float* mlp_w  = (const float*)d_in[13];
    const float* mlp_b  = (const float*)d_in[14];
    const float* mlp_g  = (const float*)d_in[15];
    const float* mlp_bt = (const float*)d_in[16];
    const float* fe_w1  = (const float*)d_in[17];
    const float* fe_b1  = (const float*)d_in[18];
    const float* fe_g1  = (const float*)d_in[19];
    const float* fe_bt1 = (const float*)d_in[20];
    const float* fe_w2  = (const float*)d_in[21];
    const float* fe_b2  = (const float*)d_in[22];
    const float* fe_g2  = (const float*)d_in[23];
    const float* fe_bt2 = (const float*)d_in[24];

    uint8_t* ws = (uint8_t*)d_ws;
    size_t off = 0;
    auto alloc = [&](size_t bytes) -> uint8_t* {
        uint8_t* p = ws + off;
        off += (bytes + 255) & ~(size_t)255;
        return p;
    };
    float*     warped  = (float*)alloc((size_t)BB * 3 * NN * 4);
    _Float16*  featT1  = (_Float16*)alloc((size_t)BB * NN * 128 * 2);
    _Float16*  featT2  = (_Float16*)alloc((size_t)BB * NN * 128 * 2);
    _Float16*  normf1  = (_Float16*)alloc((size_t)BB * NN * 128 * 2);
    _Float16*  normf2  = (_Float16*)alloc((size_t)BB * NN * 128 * 2);
    _Float16*  p11     = (_Float16*)alloc((size_t)BB * NN * 128 * 2);  // later reused as A1
    _Float16*  p22     = (_Float16*)alloc((size_t)BB * NN * 128 * 2);
    _Float16*  p12     = (_Float16*)alloc((size_t)BB * NN * 128 * 2);
    _Float16*  p21     = (_Float16*)alloc((size_t)BB * NN * 128 * 2);
    _Float16*  f1n     = (_Float16*)alloc((size_t)BB * NN * 128 * 2);
    _Float16*  f2n     = (_Float16*)alloc((size_t)BB * NN * 128 * 2);
    int*       idx1    = (int*)alloc((size_t)BB * NN * 16 * 4);
    int*       idx2    = (int*)alloc((size_t)BB * NN * 16 * 4);
    int*       idxF    = (int*)alloc((size_t)BB * NN * 16 * 4);
    _Float16*  buf0    = (_Float16*)alloc((size_t)BB * NKTOT * 160 * 2);  // t / nf_pad / u2
    _Float16*  buf1    = (_Float16*)alloc((size_t)BB * NKTOT * 128 * 2);  // u / u1
    float*     statsAll= (float*)alloc(6 * (size_t)BB * 8 * 2 * 4);
    float*     aff     = (float*)alloc((size_t)BB * 128 * 2 * 4);
    _Float16*  wt11    = (_Float16*)alloc(128 * 128 * 2);
    _Float16*  wt22    = (_Float16*)alloc(128 * 128 * 2);
    _Float16*  wmlp    = (_Float16*)alloc(128 * 128 * 2);
    _Float16*  wfe2    = (_Float16*)alloc(128 * 128 * 2);
    _Float16*  wf1a    = (_Float16*)alloc(128 * 128 * 2);
    _Float16*  wf1b    = (_Float16*)alloc(128 * 160 * 2);

    float* s0 = statsAll + 0 * BB * 16;
    float* s1 = statsAll + 1 * BB * 16;
    float* s2 = statsAll + 2 * BB * 16;
    float* s3 = statsAll + 3 * BB * 16;
    float* s4 = statsAll + 4 * BB * 16;
    float* s5 = statsAll + 5 * BB * 16;
    const float cnt = (float)(16 * NKTOT);  // elements per GroupNorm group

    hipMemsetAsync(statsAll, 0, 6 * (size_t)BB * 16 * sizeof(float), stream);

    prep_w_kernel<<<32, 256, 0, stream>>>(t11_w, t22_w, mlp_w, fe_w1, fe_w2,
                                          wt11, wt22, wmlp, wfe2, wf1a, wf1b);
    normfeat_kernel<<<BB * NN / 256, 256, 0, stream>>>(feat1, normf1);
    normfeat_kernel<<<BB * NN / 256, 256, 0, stream>>>(feat2, normf2);
    transpose_kernel<<<dim3(NN / 32, 4, BB), dim3(32, 8), 0, stream>>>(feat1, featT1);
    transpose_kernel<<<dim3(NN / 32, 4, BB), dim3(32, 8), 0, stream>>>(feat2, featT2);
    warp_kernel<<<dim3(NN / 256, BB), 256, 0, stream>>>(xyz1, xyz2, flow, warped);

    dim3 gN(NN / 64, BB), gNK(NKTOT / 64, BB);
    gconv_kernel<128><<<gN, 256, 0, stream>>>(featT1, NN, wt11, t11_b, nullptr, nullptr, p11, nullptr);
    gconv_kernel<128><<<gN, 256, 0, stream>>>(featT2, NN, wt22, t22_b, nullptr, nullptr, p22, nullptr);
    gconv_kernel<128><<<gN, 256, 0, stream>>>(featT2, NN, wt11, t11_b, nullptr, nullptr, p12, nullptr);
    gconv_kernel<128><<<gN, 256, 0, stream>>>(featT1, NN, wt22, t22_b, nullptr, nullptr, p21, nullptr);

    // ---- cross #1: (xyz1, warped, p11, p22, feat1, feat2) -> f1n
    knn_kernel<0><<<dim3(NN / 64, BB), 256, 0, stream>>>(xyz1, warped, normf1, normf2, idx1);
    build_cross_kernel<<<dim3(NN / 8, BB), 128, 0, stream>>>(xyz1, warped, p11, p22, idx1,
                                                             pos_w, pos_b, buf0, s0);
    affine_kernel<<<1, 256, 0, stream>>>(s0, gn_g, gn_b, cnt, aff);
    gconv_kernel<128><<<gNK, 256, 0, stream>>>(buf0, NKTOT, wmlp, mlp_b, aff, nullptr, buf1, s1);
    affine_kernel<<<1, 256, 0, stream>>>(s1, mlp_g, mlp_bt, cnt, aff);
    finalize_kernel<<<dim3(NN, BB), 128, 0, stream>>>(buf1, aff, f1n, nullptr, 0);

    // ---- cross #2: (warped, xyz1, p12, p21, feat2, feat1) -> f2n
    knn_kernel<0><<<dim3(NN / 64, BB), 256, 0, stream>>>(warped, xyz1, normf2, normf1, idx2);
    build_cross_kernel<<<dim3(NN / 8, BB), 128, 0, stream>>>(warped, xyz1, p12, p21, idx2,
                                                             pos_w, pos_b, buf0, s2);
    affine_kernel<<<1, 256, 0, stream>>>(s2, gn_g, gn_b, cnt, aff);
    gconv_kernel<128><<<gNK, 256, 0, stream>>>(buf0, NKTOT, wmlp, mlp_b, aff, nullptr, buf1, s3);
    affine_kernel<<<1, 256, 0, stream>>>(s3, mlp_g, mlp_bt, cnt, aff);
    finalize_kernel<<<dim3(NN, BB), 128, 0, stream>>>(buf1, aff, f2n, nullptr, 0);

    // ---- flow_embed(xyz1, warped, f1n, f2n) -> d_out
    knn_kernel<1><<<dim3(NN / 64, BB), 256, 0, stream>>>(xyz1, warped, nullptr, nullptr, idxF);
    _Float16* A1 = p11;  // p's are dead now; A1[b,n,o] = fe_w1[:,0:128] . f1n
    gconv_kernel<128><<<gN, 256, 0, stream>>>(f1n, NN, wf1a, nullptr, nullptr, nullptr, A1, nullptr);
    build_flow_kernel<<<dim3(NN / 8, BB), 160, 0, stream>>>(xyz1, warped, f2n, idxF, buf0);
    gconv_kernel<160><<<gNK, 256, 0, stream>>>(buf0, NKTOT, wf1b, fe_b1, nullptr, A1, buf1, s4);
    affine_kernel<<<1, 256, 0, stream>>>(s4, fe_g1, fe_bt1, cnt, aff);
    _Float16* u2 = buf0;  // reuse the 160-wide buffer as (B,NK,128)
    gconv_kernel<128><<<gNK, 256, 0, stream>>>(buf1, NKTOT, wfe2, fe_b2, aff, nullptr, u2, s5);
    affine_kernel<<<1, 256, 0, stream>>>(s5, fe_g2, fe_bt2, cnt, aff);
    finalize_kernel<<<dim3(NN, BB), 128, 0, stream>>>(u2, aff, nullptr, (float*)d_out, 1);
}
